// BDHPolicy_82746839924982
// MI455X (gfx1250) — matrix operand order repaired and checked by reading, not compiled, not run on hardware
//
#include <hip/hip_runtime.h>
#include <hip/hip_bf16.h>
#include <math.h>

// ---------------- problem constants ----------------
#define B_TOT   2048
#define TSTEPS  3
#define FDIM    48
#define DMODEL  128
#define NHEAD   4
#define NDIM    4096
#define ADIM    9
#define NLAYER  4
#define BPW     5            // batches per wave (15 rows + 1 pad = 16-row WMMA tile)
#define WAVES   4            // waves per workgroup (cooperative weight staging)
#define NTHREADS (WAVES * 32)

// ---- bank-conflict-free pitches (64 banks x 4B = 256B period) ----
#define XP   132             // float row pitch  (132*4  = 528B  = 4 banks skew/row)
#define BP   136             // bf16  row pitch  (136*2  = 272B  = 4 banks skew/row)
#define DP   40              // bf16  row pitch for 32-wide tiles (80B = 20 banks skew)
#define RP   33              // float row pitch for score reduction

typedef __attribute__((ext_vector_type(16))) __bf16 v16bf;
typedef __attribute__((ext_vector_type(8)))  float  v8f;

__device__ __forceinline__ __bf16 to_bf16(float f) {
  unsigned u = __float_as_uint(f);
  u += 0x7FFFu + ((u >> 16) & 1u);
  union { unsigned short s; __bf16 b; } cv;
  cv.s = (unsigned short)(u >> 16);
  return cv.b;
}

__device__ __forceinline__ v16bf make_frag(uint4 lo, uint4 hi) {
  union { uint4 q[2]; v16bf v; } u;
  u.q[0] = lo; u.q[1] = hi;
  return u.v;
}

__device__ __forceinline__ v8f wmma_bf16(v16bf a, v16bf b, v8f c) {
  return __builtin_amdgcn_wmma_f32_16x16x32_bf16(false, a, false, b, (short)0, c, false, false);
}

// A-fragment (16x32, M=row) from row-major [16][BP] bf16 tile (contiguous k = 16*hf + s).
__device__ __forceinline__ v16bf load_a_bp(const __bf16* base, int ln, int hf, int k) {
  const __bf16* p = base + ln * BP + k * 32 + hf * 16;
  return make_frag(*(const uint4*)p, *(const uint4*)(p + 8));
}

__device__ __forceinline__ float gelu_exact(float v) {
  return 0.5f * v * (1.0f + erff(v * 0.70710678118654752f));
}

#define LDS_FENCE() asm volatile("s_wait_dscnt 0" ::: "memory")
#define WAIT_ASYNC(n) asm volatile("s_wait_asynccnt " #n ::: "memory")

// CDNA5 async copy: 16B global -> LDS, tracked by ASYNCcnt, no VGPR data.
__device__ __forceinline__ void async_copy16(const __bf16* g, unsigned lds_off) {
  asm volatile("global_load_async_to_lds_b128 %0, %1, off"
               :: "v"(lds_off), "v"(g) : "memory");
}

// ---------------- weight prep: fp32 -> bf16 transposed ----------------
__global__ void bdh_prep(const float* __restrict__ enc, const float* __restrict__ encv,
                         const float* __restrict__ dec,
                         __bf16* __restrict__ encT, __bf16* __restrict__ encvT,
                         __bf16* __restrict__ decT) {
  int tid = blockIdx.x * blockDim.x + threadIdx.x;
  if (tid >= NHEAD * NDIM * DMODEL) return;
  int d0 = tid & (DMODEL - 1);
  int n  = (tid >> 7) & (NDIM - 1);
  int h  = tid >> 19;
  encT[tid]  = to_bf16(enc [(size_t)(h * DMODEL + d0) * NDIM + n]);
  encvT[tid] = to_bf16(encv[(size_t)(h * DMODEL + d0) * NDIM + n]);
  int hn = tid & (NHEAD * NDIM - 1);
  int dd = tid >> 14;
  decT[tid] = to_bf16(dec[(size_t)hn * DMODEL + dd]);
}

// ---------------- per-wave private LDS layout (31744 B / wave) ----------------
//  +0      float x[16][XP]       8448B  residual stream
//  +8448   bf16  xb[16][BP]      4352B  WMMA A operand
//  +12800  bf16  ykvb[4][16][BP] 17408B (aliased: red[32][RP] f32, ytmp[16][XP] f32)
//  +30208  bf16  stage_bf[16][DP] 1280B xy -> A-frag transpose
//  +31488  float sc[4][5][3]     +pad
#define PRIVB 31744
// staging: pass1 tile = 4 heads x [16][BP]   = 17408 B
//          pass2 tile = 2x[16][BP] + 2x[16][BP] + [128][DP] = 27648 B
#define STGB  27648

__global__ __launch_bounds__(NTHREADS)
void bdh_main(const float* __restrict__ state,
              const float* __restrict__ fe_w1, const float* __restrict__ fe_b1,
              const float* __restrict__ fe_g,  const float* __restrict__ fe_beta,
              const float* __restrict__ fe_w2, const float* __restrict__ fe_b2,
              const __bf16* __restrict__ encT, const __bf16* __restrict__ encvT,
              const __bf16* __restrict__ decT,
              const float* __restrict__ ah_w1, const float* __restrict__ ah_b1,
              const float* __restrict__ ah_w2, const float* __restrict__ ah_b2,
              const float* __restrict__ vh_w1, const float* __restrict__ vh_b1,
              const float* __restrict__ vh_w2, const float* __restrict__ vh_b2,
              const float* __restrict__ logstd,
              float* __restrict__ out_mean, float* __restrict__ out_logstd,
              float* __restrict__ out_value) {
  __shared__ __align__(16) unsigned char s_priv[WAVES][PRIVB];
  __shared__ __align__(16) unsigned char s_stg[2][STGB];

  const int tid  = threadIdx.x;
  const int wv   = tid >> 5;
  const int lane = tid & 31;
  const int ln = lane & 15;
  const int hf = lane >> 4;
  const int b0 = (blockIdx.x * WAVES + wv) * BPW;

  float*  xw   = (float*)(s_priv[wv]);
  __bf16* xbw  = (__bf16*)(s_priv[wv] + 8448);
  __bf16* ykvw = (__bf16*)(s_priv[wv] + 12800);
  float*  redw = (float*)(s_priv[wv] + 12800);   // alias (pass-1 epilogue only)
  float*  ytw  = (float*)(s_priv[wv] + 12800);   // alias (ykvb dead at use sites)
  __bf16* stbw = (__bf16*)(s_priv[wv] + 30208);
  float*  scw  = (float*)(s_priv[wv] + 31488);

  // pass1 tile: encT[4h][16n][128d] -> LDS [4][16][BP]   (1024 granules of 16B)
  auto p1_issue = [&](int ch, int bsel) {
    unsigned base = (unsigned)(unsigned long long)(void*)&s_stg[bsel][0];
    for (int g = tid; g < 1024; g += NTHREADS) {
      int h   = g >> 8;
      int nl  = (g >> 4) & 15;
      int c16 = g & 15;
      const __bf16* src = encT + ((size_t)(h * NDIM + ch * 16 + nl) * DMODEL) + c16 * 8;
      async_copy16(src, base + (unsigned)(h * 4352 + nl * 272 + c16 * 16));
    }
  };
  // pass2 tile: encT/encvT chunk pair + decT pair slice    (1536 granules of 16B)
  auto p2_issue = [&](int t, int bsel) {
    int h = t >> 7, pr = t & 127;
    unsigned base = (unsigned)(unsigned long long)(void*)&s_stg[bsel][0];
    for (int g = tid; g < 1536; g += NTHREADS) {
      const __bf16* src; unsigned dst;
      if (g < 1024) {
        const __bf16* W = (g < 512) ? encT : encvT;
        int gg  = g & 511;
        int cc  = gg >> 8;
        int nl  = (gg >> 4) & 15;
        int c16 = gg & 15;
        src = W + ((size_t)(h * NDIM + (pr * 2 + cc) * 16 + nl) * DMODEL) + c16 * 8;
        dst = base + (unsigned)((g < 512 ? 0 : 8704) + cc * 4352 + nl * 272 + c16 * 16);
      } else {
        int q = g - 1024;
        int dcol = q >> 2;
        int c16  = q & 3;
        src = decT + (size_t)dcol * (NHEAD * NDIM) + h * NDIM + pr * 32 + c16 * 8;
        dst = base + (unsigned)(17408 + dcol * 80 + c16 * 16);
      }
      async_copy16(src, dst);
    }
  };

  // ================= frame encoder (VALU, per-wave private) ====================
  for (int idx = lane; idx < 16 * DMODEL; idx += 32) {
    int m = idx >> 7, j = idx & 127;
    float s = 0.f;
    if (m < 15) {
      int b = b0 + m / 3; if (b >= B_TOT) b = B_TOT - 1;
      int t = m % 3;
      const float* sp = state + (size_t)b * (TSTEPS * FDIM) + t * FDIM;
      s = fe_b1[j];
      for (int f = 0; f < FDIM; ++f) s = fmaf(sp[f], fe_w1[f * DMODEL + j], s);
    }
    ytw[m * XP + j] = s;
  }
  LDS_FENCE();
  {
    int c0 = hf * 64;
    float s1 = 0.f, s2 = 0.f;
    for (int c = 0; c < 64; ++c) { float v = ytw[ln * XP + c0 + c]; s1 += v; s2 += v * v; }
    s1 += __shfl_xor(s1, 16, 32); s2 += __shfl_xor(s2, 16, 32);
    float mean = s1 * (1.f / 128.f);
    float rstd = rsqrtf(s2 * (1.f / 128.f) - mean * mean + 1e-5f);
    for (int c = 0; c < 64; ++c) {
      float v = (ytw[ln * XP + c0 + c] - mean) * rstd * fe_g[c0 + c] + fe_beta[c0 + c];
      ytw[ln * XP + c0 + c] = gelu_exact(v);
    }
  }
  LDS_FENCE();
  for (int idx = lane; idx < 16 * DMODEL; idx += 32) {
    int m = idx >> 7, j = idx & 127;
    float s = fe_b2[j];
    for (int k = 0; k < DMODEL; ++k) s = fmaf(ytw[m * XP + k], fe_w2[k * DMODEL + j], s);
    xw[m * XP + j] = (m < 15) ? s : 0.f;
  }
  LDS_FENCE();
  {
    int c0 = hf * 64;
    float s1 = 0.f, s2 = 0.f;
    for (int c = 0; c < 64; ++c) { float v = xw[ln * XP + c0 + c]; s1 += v; s2 += v * v; }
    s1 += __shfl_xor(s1, 16, 32); s2 += __shfl_xor(s2, 16, 32);
    float mean = s1 * (1.f / 128.f);
    float rstd = rsqrtf(s2 * (1.f / 128.f) - mean * mean + 1e-5f);
    for (int c = 0; c < 64; ++c) {
      float v = (xw[ln * XP + c0 + c] - mean) * rstd;
      xw[ln * XP + c0 + c] = v;
      xbw[ln * BP + c0 + c] = to_bf16(v);
    }
  }
  LDS_FENCE();

  // ================= transformer layers =================
  for (int layer = 0; layer < NLAYER; ++layer) {
    v16bf axf[4];
#pragma unroll
    for (int k = 0; k < 4; ++k) axf[k] = load_a_bp(xbw, ln, hf, k);

    // ---- pass 1: stream N; register-resident causal scores ----
    float scacc[4][8];
#pragma unroll
    for (int h = 0; h < 4; ++h)
#pragma unroll
      for (int s = 0; s < 8; ++s) scacc[h][s] = 0.f;

    p1_issue(0, 0);
    p1_issue(1, 1);
    for (int ti = 0; ti < 256; ++ti) {
      if (ti + 1 < 256) { WAIT_ASYNC(8); } else { WAIT_ASYNC(0); }
      __syncthreads();
      const __bf16* tile = (const __bf16*)&s_stg[ti & 1][0];
      // RoPE factors: theta=2^16 -> freq = 2^(-n/256); double angle by identity.
      int nl = ti * 16 + ln;
      float f   = exp2f((float)nl * (-1.0f / 256.0f));
      float s1v = __sinf(f), c1 = __cosf(f);
      float s2v = 2.f * s1v * c1;
      float c2  = 1.f - 2.f * s1v * s1v;
      // per-row cos/sin (t = row%3), hoisted out of the head loop
      float cvr[8], svr[8];
      if (hf == 0) {   // rows 0..7 : t = 0,1,2,0,1,2,0,1
        cvr[0]=1.f; svr[0]=0.f;  cvr[1]=c1; svr[1]=s1v; cvr[2]=c2; svr[2]=s2v;
        cvr[3]=1.f; svr[3]=0.f;  cvr[4]=c1; svr[4]=s1v; cvr[5]=c2; svr[5]=s2v;
        cvr[6]=1.f; svr[6]=0.f;  cvr[7]=c1; svr[7]=s1v;
      } else {         // rows 8..15: t = 2,0,1,2,0,1,2,0
        cvr[0]=c2;  svr[0]=s2v;  cvr[1]=1.f; svr[1]=0.f; cvr[2]=c1; svr[2]=s1v;
        cvr[3]=c2;  svr[3]=s2v;  cvr[4]=1.f; svr[4]=0.f; cvr[5]=c1; svr[5]=s1v;
        cvr[6]=c2;  svr[6]=s2v;  cvr[7]=1.f; svr[7]=0.f;
      }
#pragma unroll
      for (int h = 0; h < NHEAD; ++h) {
        v8f acc = {};
#pragma unroll
        for (int k = 0; k < 4; ++k) {
          const __bf16* p = tile + h * 2176 + ln * BP + k * 32 + hf * 16;
          acc = wmma_bf16(axf[k], make_frag(*(const uint4*)p, *(const uint4*)(p + 8)), acc);
        }
        float q[8];
#pragma unroll
        for (int r = 0; r < 8; ++r) {
          float v = fmaxf(acc[r], 0.f);
          float prt = __shfl_xor(v, 1, 32);
          float rot = (ln & 1) ? prt : -prt;
          q[r] = v * cvr[r] + rot * svr[r];
        }
        float q6o = __shfl_xor(q[6], 16, 32);
        float q7o = __shfl_xor(q[7], 16, 32);
        if (hf == 0) {          // batches 0,1 + (7,6) of batch 2
          scacc[h][0] += q[1] * q[0]; scacc[h][1] += q[2] * q[0]; scacc[h][2] += q[2] * q[1];
          scacc[h][3] += q[4] * q[3]; scacc[h][4] += q[5] * q[3]; scacc[h][5] += q[5] * q[4];
          scacc[h][6] += q[7] * q[6];
        } else {                // (8,6),(8,7) of batch 2 + batches 3,4
          scacc[h][0] += q[0] * q6o;  scacc[h][1] += q[0] * q7o;
          scacc[h][2] += q[2] * q[1]; scacc[h][3] += q[3] * q[1]; scacc[h][4] += q[3] * q[2];
          scacc[h][5] += q[5] * q[4]; scacc[h][6] += q[6] * q[4]; scacc[h][7] += q[6] * q[5];
        }
      }
      __syncthreads();
      if (ti + 2 < 256) p1_issue(ti + 2, ti & 1);
    }
    // pass-2 prefetch overlaps the private score/ykv epilogue
    p2_issue(0, 0);
    p2_issue(1, 1);

    // ---- reduce scores across the 16 n-lanes (once per layer) ----
#pragma unroll
    for (int h = 0; h < 4; ++h)
#pragma unroll
      for (int s = 0; s < 8; ++s)
        redw[(hf * 16 + ln) * RP + h * 8 + s] = scacc[h][s];
    LDS_FENCE();
    for (int idx = lane; idx < 60; idx += 32) {
      int h = idx / 15, combo = idx % 15;
      int srchf = (combo < 7) ? 0 : 1;
      int slot  = (combo < 7) ? combo : combo - 7;
      float s = 0.f;
      for (int l2 = 0; l2 < 16; ++l2)
        s += redw[(srchf * 16 + l2) * RP + h * 8 + slot];
      scw[(h * BPW + combo / 3) * 3 + combo % 3] = s;
    }
    LDS_FENCE();

    // ---- ykv = LN(scores @ x) ----
    for (int idx = lane; idx < NHEAD * 16; idx += 32) {
      int h = idx >> 4, m = idx & 15;
      float w0 = 0.f, w1 = 0.f; int r0 = 0, r1 = 0;
      if (m < 15) {
        int bi = m / 3, tt = m % 3;
        r0 = 3 * bi; r1 = 3 * bi + 1;
        if (tt == 1)      { w0 = scw[(h * BPW + bi) * 3 + 0]; }
        else if (tt == 2) { w0 = scw[(h * BPW + bi) * 3 + 1]; w1 = scw[(h * BPW + bi) * 3 + 2]; }
      }
      float s1 = 0.f, s2 = 0.f;
      for (int d0 = 0; d0 < DMODEL; ++d0) {
        float v = w0 * xw[r0 * XP + d0] + w1 * xw[r1 * XP + d0];
        s1 += v; s2 += v * v;
      }
      float mean = s1 * (1.f / 128.f);
      float rstd = rsqrtf(s2 * (1.f / 128.f) - mean * mean + 1e-5f);
      for (int d0 = 0; d0 < DMODEL; ++d0) {
        float v = w0 * xw[r0 * XP + d0] + w1 * xw[r1 * XP + d0];
        ykvw[(h * 16 + m) * BP + d0] = to_bf16((v - mean) * rstd);
      }
    }
    LDS_FENCE();

    // ---- pass 2: recompute x_sparse, y_sparse, xy; y += xy @ decoder ----
    v8f yacc[8];
#pragma unroll
    for (int nt = 0; nt < 8; ++nt) yacc[nt] = (v8f){};
    v16bf ayf[4];
    for (int ti = 0; ti < 512; ++ti) {
      if ((ti & 127) == 0) {
        int h = ti >> 7;
#pragma unroll
        for (int k = 0; k < 4; ++k) ayf[k] = load_a_bp(ykvw + h * 16 * BP, ln, hf, k);
      }
      if (ti + 1 < 512) { WAIT_ASYNC(12); } else { WAIT_ASYNC(0); }
      __syncthreads();
      const __bf16* xS = (const __bf16*)&s_stg[ti & 1][0];
      const __bf16* vS = xS + 4352;
      const __bf16* dS = xS + 8704;
#pragma unroll
      for (int cc = 0; cc < 2; ++cc) {
        v8f ax = {}; v8f ay = {};
#pragma unroll
        for (int k = 0; k < 4; ++k) {
          const __bf16* p = xS + cc * 2176 + ln * BP + k * 32 + hf * 16;
          ax = wmma_bf16(axf[k], make_frag(*(const uint4*)p, *(const uint4*)(p + 8)), ax);
        }
#pragma unroll
        for (int k = 0; k < 4; ++k) {
          const __bf16* p = vS + cc * 2176 + ln * BP + k * 32 + hf * 16;
          ay = wmma_bf16(ayf[k], make_frag(*(const uint4*)p, *(const uint4*)(p + 8)), ay);
        }
#pragma unroll
        for (int r = 0; r < 8; ++r)
          stbw[(r + 8 * hf) * DP + cc * 16 + ln] = to_bf16(fmaxf(ax[r], 0.f) * fmaxf(ay[r], 0.f));
      }
      LDS_FENCE();
      v16bf axy;
      {
        const __bf16* p0 = stbw + ln * DP + hf * 16;
        const __bf16* p1 = stbw + ln * DP + hf * 16 + 8;
        axy = make_frag(*(const uint4*)p0, *(const uint4*)p1);
      }
#pragma unroll
      for (int nt = 0; nt < 8; ++nt) {
        const __bf16* p = dS + (nt * 16 + ln) * DP + hf * 16;
        yacc[nt] = wmma_bf16(axy, make_frag(*(const uint4*)p, *(const uint4*)(p + 8)), yacc[nt]);
      }
      __syncthreads();
      if (ti + 2 < 512) p2_issue(ti + 2, ti & 1);
    }

    // ---- x = LN(x + LN(y))  (private; ytmp aliases dead ykvb) ----
#pragma unroll
    for (int nt = 0; nt < 8; ++nt)
#pragma unroll
      for (int r = 0; r < 8; ++r)
        ytw[(r + 8 * hf) * XP + nt * 16 + ln] = yacc[nt][r];
    LDS_FENCE();
    {
      int c0 = hf * 64;
      float s1 = 0.f, s2 = 0.f;
      for (int c = 0; c < 64; ++c) { float v = ytw[ln * XP + c0 + c]; s1 += v; s2 += v * v; }
      s1 += __shfl_xor(s1, 16, 32); s2 += __shfl_xor(s2, 16, 32);
      float mean = s1 * (1.f / 128.f);
      float rstd = rsqrtf(s2 * (1.f / 128.f) - mean * mean + 1e-5f);
      float t1 = 0.f, t2 = 0.f;
      for (int c = 0; c < 64; ++c) {
        float z = xw[ln * XP + c0 + c] + (ytw[ln * XP + c0 + c] - mean) * rstd;
        ytw[ln * XP + c0 + c] = z; t1 += z; t2 += z * z;
      }
      t1 += __shfl_xor(t1, 16, 32); t2 += __shfl_xor(t2, 16, 32);
      float mean2 = t1 * (1.f / 128.f);
      float rstd2 = rsqrtf(t2 * (1.f / 128.f) - mean2 * mean2 + 1e-5f);
      for (int c = 0; c < 64; ++c) {
        float z = (ytw[ln * XP + c0 + c] - mean2) * rstd2;
        xw[ln * XP + c0 + c] = z;
        xbw[ln * BP + c0 + c] = to_bf16(z);
      }
    }
    LDS_FENCE();
  }

  // ================= policy / value heads =================
  for (int idx = lane; idx < BPW * DMODEL; idx += 32) {
    int bi = idx >> 7, j = idx & 127;
    const float* xr = xw + (3 * bi + 2) * XP;
    float sa = ah_b1[j], sv = vh_b1[j];
    for (int d0 = 0; d0 < DMODEL; ++d0) {
      float xv = xr[d0];
      sa = fmaf(xv, ah_w1[d0 * DMODEL + j], sa);
      sv = fmaf(xv, vh_w1[d0 * DMODEL + j], sv);
    }
    ytw[bi * XP + j]       = gelu_exact(sa);
    ytw[(8 + bi) * XP + j] = gelu_exact(sv);
  }
  LDS_FENCE();
  for (int idx = lane; idx < BPW * ADIM; idx += 32) {
    int bi = idx / ADIM, k = idx % ADIM;
    int b = b0 + bi;
    if (b < B_TOT) {
      float s = ah_b2[k];
      for (int j = 0; j < DMODEL; ++j) s = fmaf(ytw[bi * XP + j], ah_w2[j * ADIM + k], s);
      out_mean[(size_t)b * ADIM + k] = s;
      out_logstd[(size_t)b * ADIM + k] = fminf(fmaxf(logstd[k], -5.f), 2.f);
    }
  }
  for (int idx = lane; idx < BPW; idx += 32) {
    int b = b0 + idx;
    if (b < B_TOT) {
      float s = vh_b2[0];
      for (int j = 0; j < DMODEL; ++j) s = fmaf(ytw[(8 + idx) * XP + j], vh_w2[j], s);
      out_value[b] = s;
    }
  }
}

// ---------------- host launcher ----------------
extern "C" void kernel_launch(void* const* d_in, const int* in_sizes, int n_in,
                              void* d_out, int out_size, void* d_ws, size_t ws_size,
                              hipStream_t stream) {
  (void)in_sizes; (void)n_in; (void)out_size; (void)ws_size;
  const float* state   = (const float*)d_in[0];
  const float* fe_w1   = (const float*)d_in[1];
  const float* fe_b1   = (const float*)d_in[2];
  const float* fe_g    = (const float*)d_in[3];
  const float* fe_beta = (const float*)d_in[4];
  const float* fe_w2   = (const float*)d_in[5];
  const float* fe_b2   = (const float*)d_in[6];
  const float* enc     = (const float*)d_in[7];
  const float* encv    = (const float*)d_in[8];
  const float* dec     = (const float*)d_in[9];
  const float* ah_w1   = (const float*)d_in[10];
  const float* ah_b1   = (const float*)d_in[11];
  const float* ah_w2   = (const float*)d_in[12];
  const float* ah_b2   = (const float*)d_in[13];
  const float* vh_w1   = (const float*)d_in[14];
  const float* vh_b1   = (const float*)d_in[15];
  const float* vh_w2   = (const float*)d_in[16];
  const float* vh_b2   = (const float*)d_in[17];
  const float* logstd  = (const float*)d_in[18];

  const int WELEMS = NHEAD * NDIM * DMODEL;
  __bf16* encT  = (__bf16*)d_ws;
  __bf16* encvT = encT + WELEMS;
  __bf16* decT  = encvT + WELEMS;

  float* o = (float*)d_out;
  float* out_mean   = o;
  float* out_logstd = o + (size_t)B_TOT * ADIM;
  float* out_value  = o + 2 * (size_t)B_TOT * ADIM;

  bdh_prep<<<(WELEMS + 255) / 256, 256, 0, stream>>>(enc, encv, dec, encT, encvT, decT);

  int nblocks = (B_TOT + BPW * WAVES - 1) / (BPW * WAVES);   // 103 cooperative WGs
  bdh_main<<<nblocks, NTHREADS, 0, stream>>>(state, fe_w1, fe_b1, fe_g, fe_beta, fe_w2, fe_b2,
                                             encT, encvT, decT,
                                             ah_w1, ah_b1, ah_w2, ah_b2,
                                             vh_w1, vh_b1, vh_w2, vh_b2, logstd,
                                             out_mean, out_logstd, out_value);
}
